// OFT_63050119905554
// MI455X (gfx1250) — compile-verified, matrix operand order checked by reading, not run on hardware
//
#include <hip/hip_runtime.h>

// ---------------- problem constants ----------------
#define WF    320
#define HF    96
#define CCH   256
#define BATCH 2
#define NYJ   7
#define NPOS  16384            // 128*128
#define KTOT  1792             // CCH*NYJ
#define NOUT  256

// output flat offsets (floats)
#define O_ORTHO 0
#define O_II    8388608
#define O_GTL   24117248
#define G_SZ    458752

// GEMM tiling
#define MTILE  32
#define CCHUNK 32
#define KCH    (CCHUNK*NYJ)    // 224 = 7 WMMA k-steps
#define LSTR   228             // padded LDS stride (halves)

typedef __attribute__((ext_vector_type(16))) __bf16 v16bf;
typedef __attribute__((ext_vector_type(8)))  float  v8f;

struct Tap { int off; float w; };

// ---------------- w -> bf16 ----------------
__global__ void k_wbf(const float* __restrict__ w, unsigned short* __restrict__ wbf, int n) {
  int i = blockIdx.x * blockDim.x + threadIdx.x;
  if (i < n) {
    __bf16 h = (__bf16)w[i];
    wbf[i] = __builtin_bit_cast(unsigned short, h);
  }
}

// ---------------- row cumsum (axis W): features -> ii ----------------
__global__ void k_rowscan(const float* __restrict__ f, float* __restrict__ ii) {
  int lane = threadIdx.x & 31;
  int wv   = threadIdx.x >> 5;
  int row  = blockIdx.x * 8 + wv;               // over B*C*H rows
  const float* src = f  + (size_t)row * WF;
  float*       dst = ii + (size_t)row * WF;
  float carry = 0.f;
  #pragma unroll
  for (int ch = 0; ch < WF / 32; ++ch) {
    float x = src[ch * 32 + lane];
    #pragma unroll
    for (int off = 1; off < 32; off <<= 1) {
      float y = __shfl_up(x, off, 32);
      if (lane >= off) x += y;
    }
    x += carry;
    dst[ch * 32 + lane] = x;
    carry = __shfl(x, 31, 32);
  }
}

// ---------------- column cumsum (axis H) in place ----------------
__global__ void k_colscan(float* __restrict__ ii) {
  int idx = blockIdx.x * blockDim.x + threadIdx.x;  // over B*C*W
  if (idx >= BATCH * CCH * WF) return;
  int bc = idx / WF, wc = idx % WF;
  float* p = ii + (size_t)bc * (HF * WF) + wc;
  float s = p[0];
  for (int y = 1; y < HF; ++y) { s += p[y * WF]; p[y * WF] = s; }
}

// ---------------- geometry: bbox -> g_* outputs + tap table ----------------
__device__ inline float2 nc_point(const float* cb, const float* gb, int dI, int wI, float yoff) {
  const float* g = gb + ((size_t)dI * 129 + wI) * 3;
  float cx = g[0], cy = g[1] + yoff, cz = g[2];
  float h0 = fmaf(cb[0], cx, fmaf(cb[1], cy, fmaf(cb[2],  cz, cb[3])));
  float h1 = fmaf(cb[4], cx, fmaf(cb[5], cy, fmaf(cb[6],  cz, cb[7])));
  float h2 = fmaf(cb[8], cx, fmaf(cb[9], cy, fmaf(cb[10], cz, cb[11])));
  float ix = h0 / h2, iy = h1 / h2;
  float nx = fminf(fmaxf(ix * (2.0f / 2560.0f) - 1.0f, -1.0f), 1.0f);
  float ny = fminf(fmaxf(iy * (2.0f / 768.0f)  - 1.0f, -1.0f), 1.0f);
  return make_float2(nx, ny);
}

__global__ void k_geom(const float* __restrict__ calib, const float* __restrict__ grid,
                       float* __restrict__ gtl, float* __restrict__ gbr,
                       float* __restrict__ gtr, float* __restrict__ gbl,
                       Tap* __restrict__ taps) {
  int idx = blockIdx.x * blockDim.x + threadIdx.x;   // B*7*16384
  if (idx >= BATCH * NYJ * NPOS) return;
  int pos = idx % NPOS;
  int j   = (idx / NPOS) % NYJ;
  int b   = idx / (NPOS * NYJ);
  int d = pos >> 7, wd = pos & 127;
  const float* cb = calib + b * 12;
  const float* gb = grid + (size_t)b * 129 * 129 * 3;
  float y0 = j * 0.5f - 2.0f, y1 = (j + 1) * 0.5f - 2.0f;
  float2 p00 = nc_point(cb, gb, d,     wd,     y0);
  float2 p01 = nc_point(cb, gb, d + 1, wd,     y0);
  float2 p10 = nc_point(cb, gb, d + 1, wd + 1, y1);
  float2 p11 = nc_point(cb, gb, d,     wd + 1, y1);
  float minx = fminf(p00.x, p01.x), miny = fminf(p00.y, p01.y);
  float maxx = fmaxf(p10.x, p11.x), maxy = fmaxf(p10.y, p11.y);
  gtl[idx*2] = minx; gtl[idx*2+1] = miny;
  gbr[idx*2] = maxx; gbr[idx*2+1] = maxy;
  gtr[idx*2] = maxx; gtr[idx*2+1] = miny;
  gbl[idx*2] = minx; gbl[idx*2+1] = maxy;
  float area = (maxx - minx) * (maxy - miny) * (HF * WF * 0.25f) + 1e-6f;
  float inv  = (area > 1e-6f) ? (1.0f / area) : 0.0f;   // visible/area folded
  Tap* tp = taps + (size_t)idx * 16;
  float gx[4] = {minx, maxx, maxx, minx};
  float gy[4] = {miny, maxy, miny, maxy};
  float sg[4] = {inv, inv, -inv, -inv};                 // tl + br - tr - bl
  #pragma unroll
  for (int cI = 0; cI < 4; ++cI) {
    float x = (gx[cI] + 1.0f) * (WF * 0.5f) - 0.5f;
    float y = (gy[cI] + 1.0f) * (HF * 0.5f) - 0.5f;
    float xf = floorf(x), yf = floorf(y);
    float wx = x - xf, wy = y - yf;
    int xi = (int)xf, yi = (int)yf;
    float wts[4] = {(1-wx)*(1-wy), wx*(1-wy), (1-wx)*wy, wx*wy};
    const int dx[4] = {0,1,0,1}, dy[4] = {0,0,1,1};
    #pragma unroll
    for (int t = 0; t < 4; ++t) {
      int xt = xi + dx[t], yt = yi + dy[t];
      bool valid = (xt >= 0) && (xt < WF) && (yt >= 0) && (yt < HF);
      int xc = min(max(xt, 0), WF - 1), yc = min(max(yt, 0), HF - 1);
      tp[cI*4 + t].off = yc * WF + xc;
      tp[cI*4 + t].w   = valid ? sg[cI] * wts[t] : 0.0f;
    }
  }
}

// ---------------- fused vox(LDS) + bf16 WMMA GEMM + bias + relu ----------------
__global__ __launch_bounds__(256) void k_gemm(const float* __restrict__ ii,
                                              const unsigned short* __restrict__ wbf_u,
                                              const Tap* __restrict__ taps,
                                              const float* __restrict__ bias,
                                              float* __restrict__ ortho) {
  __shared__ __bf16 voxLDS[MTILE * LSTR];
  const __bf16* wbf = (const __bf16*)wbf_u;
  int tid  = threadIdx.x;
  int lane = tid & 31, wv = tid >> 5;
  int m0      = blockIdx.x * MTILE;
  int b       = m0 / NPOS;
  int posbase = m0 % NPOS;

  v8f acc00 = {}, acc01 = {}, acc10 = {}, acc11 = {};
  int n0    = wv * 32;                    // each wave: n-tiles n0, n0+16
  int lo    = (lane < 16) ? 0 : 8;        // 16-bit fragment k-offset per lane half
  int rA    = lane & 15;
  int ncol0 = n0 + (lane & 15);
  int ncol1 = ncol0 + 16;

  // phase-1 mapping: (row, j) pair per thread; taps reused across all 256 channels
  int  pr   = tid / NYJ;
  int  pj   = tid % NYJ;
  bool pact = tid < MTILE * NYJ;          // 224 active producers
  Tap tp[16];
  if (pact) {
    const Tap* ts = taps + (((size_t)(b * NYJ + pj)) * NPOS + (posbase + pr)) * 16;
    #pragma unroll
    for (int t = 0; t < 16; ++t) tp[t] = ts[t];
  }

  for (int ch = 0; ch < CCH / CCHUNK; ++ch) {
    if (pact) {
      #pragma unroll 4
      for (int cl = 0; cl < CCHUNK; ++cl) {
        int c = ch * CCHUNK + cl;
        const float* iiC = ii + ((size_t)(b * CCH + c)) * (HF * WF);
        float a = 0.f;
        #pragma unroll
        for (int t = 0; t < 16; ++t) a = fmaf(tp[t].w, iiC[tp[t].off], a);
        voxLDS[pr * LSTR + cl * NYJ + pj] = (__bf16)a;   // k_local = cl*7 + pj
      }
    }
    __syncthreads();
    #pragma unroll
    for (int ks = 0; ks < KCH / 32; ++ks) {              // 7 WMMA k-steps
      int kb = ks * 32;
      v16bf a0, a1, b0, b1;
      const __bf16* ap0 = &voxLDS[rA * LSTR + kb + lo];
      const __bf16* ap1 = &voxLDS[(rA + 16) * LSTR + kb + lo];
      const __bf16* bp0 = wbf + (size_t)ncol0 * KTOT + ch * KCH + kb + lo;
      const __bf16* bp1 = wbf + (size_t)ncol1 * KTOT + ch * KCH + kb + lo;
      #pragma unroll
      for (int i = 0; i < 8; ++i) {
        a0[i] = ap0[i]; a0[i + 8] = ap0[16 + i];
        a1[i] = ap1[i]; a1[i + 8] = ap1[16 + i];
        b0[i] = bp0[i]; b0[i + 8] = bp0[16 + i];
        b1[i] = bp1[i]; b1[i + 8] = bp1[16 + i];
      }
      acc00 = __builtin_amdgcn_wmma_f32_16x16x32_bf16(false, a0, false, b0, (short)0, acc00, false, false);
      acc01 = __builtin_amdgcn_wmma_f32_16x16x32_bf16(false, a0, false, b1, (short)0, acc01, false, false);
      acc10 = __builtin_amdgcn_wmma_f32_16x16x32_bf16(false, a1, false, b0, (short)0, acc10, false, false);
      acc11 = __builtin_amdgcn_wmma_f32_16x16x32_bf16(false, a1, false, b1, (short)0, acc11, false, false);
    }
    __syncthreads();
  }

  // epilogue: bias + relu, scatter to (B, N, D, Wd)
  int nlo = lane & 15;
  int mhi = (lane < 16) ? 0 : 8;
  float bias0 = bias[n0 + nlo];
  float bias1 = bias[n0 + 16 + nlo];
  #pragma unroll
  for (int r = 0; r < 8; ++r) {
    int mA = posbase + r + mhi;
    int mB = posbase + 16 + r + mhi;
    ortho[((size_t)(b * NOUT + n0 + nlo))      * NPOS + mA] = fmaxf(acc00[r] + bias0, 0.f);
    ortho[((size_t)(b * NOUT + n0 + 16 + nlo)) * NPOS + mA] = fmaxf(acc01[r] + bias1, 0.f);
    ortho[((size_t)(b * NOUT + n0 + nlo))      * NPOS + mB] = fmaxf(acc10[r] + bias0, 0.f);
    ortho[((size_t)(b * NOUT + n0 + 16 + nlo)) * NPOS + mB] = fmaxf(acc11[r] + bias1, 0.f);
  }
}

extern "C" void kernel_launch(void* const* d_in, const int* in_sizes, int n_in,
                              void* d_out, int out_size, void* d_ws, size_t ws_size,
                              hipStream_t stream) {
  const float* features = (const float*)d_in[0];
  const float* calib    = (const float*)d_in[1];
  const float* grid     = (const float*)d_in[2];
  const float* w        = (const float*)d_in[3];
  const float* bias     = (const float*)d_in[4];
  float* out = (float*)d_out;
  float* ortho = out + O_ORTHO;
  float* ii    = out + O_II;
  float* gtl   = out + O_GTL;
  float* gbr   = gtl + G_SZ;
  float* gtr   = gbr + G_SZ;
  float* gbl   = gtr + G_SZ;
  unsigned short* wbf = (unsigned short*)d_ws;                  // 917,504 B
  Tap* taps = (Tap*)((char*)d_ws + (1 << 20));                  // 29.4 MB

  k_wbf    <<<(KTOT * NOUT + 255) / 256, 256, 0, stream>>>(w, wbf, KTOT * NOUT);
  k_rowscan<<<BATCH * CCH * HF / 8,       256, 0, stream>>>(features, ii);
  k_colscan<<<(BATCH * CCH * WF + 255) / 256, 256, 0, stream>>>(ii);
  k_geom   <<<(BATCH * NYJ * NPOS + 255) / 256, 256, 0, stream>>>(calib, grid, gtl, gbr, gtr, gbl, taps);
  k_gemm   <<<(BATCH * NPOS) / MTILE,     256, 0, stream>>>(ii, wbf, taps, bias, ortho);
}
